// FNCModel_45045617000622
// MI455X (gfx1250) — matrix operand order repaired and checked by reading, not compiled, not run on hardware
//
#include <hip/hip_runtime.h>
#include <cstdint>
#include <cstddef>

// ---------------------------------------------------------------------------
// FNC model: embeddings -> 2x GRU(H=1024) -> concat -> linear(2048->4)
// bf16 WMMA (v_wmma_f32_16x16x32_bf16) for all GEMMs, f32 state/activations.
// Weights staged block-wide through LDS with GLOBAL_LOAD_ASYNC_TO_LDS_B128
// (ASYNCcnt), double-buffered against the WMMA pipe.
// ---------------------------------------------------------------------------

#define Bz   128      // batch
#define Hh   1024     // hidden
#define Ee   300      // embedding dim
#define EP   320      // padded embedding dim (multiple of 32)
#define KT   1344     // EP + H  (padded concat dim, 42 * 32)
#define KCH  42       // KT / 32
#define NG   2048     // 2*H (gates width)
#define HLEN 32       // headline T
#define BLEN 800      // body T
#define NCLS 4

typedef __attribute__((ext_vector_type(16))) __bf16 v16bf;
typedef __attribute__((ext_vector_type(8)))  float  v8f;

union FragBF { v16bf v; uint4 q[2]; };

__device__ __forceinline__ unsigned short f2bf(float f) {
    union { float f; unsigned u; } v; v.f = f;
    unsigned x = v.u + 0x7fffu + ((v.u >> 16) & 1u);   // round-to-nearest-even
    return (unsigned short)(x >> 16);
}

// CDNA5 async global->LDS copy (16B per lane), tracked by ASYNCcnt.
#define ASYNC_COPY_B128(lds_u32, gaddr64)                                   \
    asm volatile("global_load_async_to_lds_b128 %0, %1, off"                \
                 :: "v"(lds_u32), "v"(gaddr64) : "memory")
#define WAIT_ASYNC0() asm volatile("s_wait_asynccnt 0x0" ::: "memory")

// ---------------- precompute kernels ----------------------------------------

// W [D=1324][N] f32  ->  Wt [N][KT] bf16, rows 300..319 zero-padded.
__global__ void convert_weight(const float* __restrict__ W,
                               unsigned short* __restrict__ Wt, int N) {
    int idx = blockIdx.x * blockDim.x + threadIdx.x;
    if (idx >= N * KT) return;
    int n = idx / KT, k = idx % KT;
    float v;
    if (k < Ee)        v = W[(size_t)k * N + n];
    else if (k < EP)   v = 0.0f;
    else               v = W[(size_t)(Ee + (k - EP)) * N + n];
    Wt[idx] = f2bf(v);
}

// tokens [B][T][1] -> xp [T][B][EP] bf16 (zero padded cols E..EP)
__global__ void gather_embed(const int* __restrict__ toks, int T,
                             const float* __restrict__ emb,
                             unsigned short* __restrict__ xp, int total) {
    int idx = blockIdx.x * blockDim.x + threadIdx.x;
    if (idx >= total) return;
    int t = idx / (Bz * EP);
    int r = idx % (Bz * EP);
    int b = r / EP, k = r % EP;
    int tok = toks[(size_t)b * T + t];
    float v = (k < Ee) ? emb[(size_t)tok * Ee + k] : 0.0f;
    xp[idx] = f2bf(v);
}

// TF seq-length: used[b,t] = sign(max_e |x[b,t,e]|)
__global__ void seq_used(const int* __restrict__ toks, int T,
                         const float* __restrict__ emb, int* __restrict__ used) {
    int idx = blockIdx.x * blockDim.x + threadIdx.x;
    if (idx >= Bz * T) return;
    int b = idx / T, t = idx % T;
    int tok = toks[(size_t)b * T + t];
    float m = 0.0f;
    for (int e = 0; e < Ee; ++e) m = fmaxf(m, fabsf(emb[(size_t)tok * Ee + e]));
    used[idx] = (m > 0.0f) ? 1 : 0;
}

__global__ void seq_len(const int* __restrict__ used, int T, int* __restrict__ sl) {
    int b = threadIdx.x;
    if (b >= Bz) return;
    int s = 0;
    for (int t = 0; t < T; ++t) s += used[(size_t)b * T + t];
    sl[b] = s;
}

__global__ void zero_state(float* __restrict__ hf, unsigned short* __restrict__ hbf) {
    int idx = blockIdx.x * blockDim.x + threadIdx.x;
    if (idx >= Bz * Hh) return;
    hf[idx] = 0.0f;
    hbf[idx] = 0;
}

// ---------------- shared GEMM core ------------------------------------------
// Block = 8 waves, one ngroup (64 N-cols) per block, one M-tile per wave.
// Per K-chunk (32): 4KB weight slice staged via async copy to LDS (double
// buffered); A fragment double-buffered in registers; 4 WMMAs per wave.
__device__ __forceinline__ void gru_gemm(const unsigned short* __restrict__ xrow,
                                         const unsigned short* __restrict__ hrow,
                                         const unsigned short* __restrict__ Wt,
                                         int n0base,
                                         unsigned short (*smW)[64][32],  // [2][64][32]
                                         v8f* acc) {
    int tid  = threadIdx.x;
    int lane = tid & 31;
    int lo = lane & 15, hi = lane >> 4;

    // copier role: thread -> (column, 8-elem group) of the 64x32 slice
    int cc = tid >> 2;   // 0..63
    int cj = tid & 3;    // 0..3
    const unsigned short* gsrc = Wt + (size_t)(n0base + cc) * KT + cj * 8;
    unsigned ldsd0 = (unsigned)(uintptr_t)&smW[0][cc][cj * 8];
    unsigned ldsd1 = (unsigned)(uintptr_t)&smW[1][cc][cj * 8];

    // prologue: stage chunk 0, preload A chunk 0
    ASYNC_COPY_B128(ldsd0, (unsigned long long)(uintptr_t)gsrc);
    FragBF a_cur, a_nxt;
    a_cur.q[0] = *(const uint4*)(xrow + hi * 16);
    a_cur.q[1] = *(const uint4*)(xrow + hi * 16 + 8);
    WAIT_ASYNC0();
    __syncthreads();

    for (int kc = 0; kc < KCH; ++kc) {
        int sel = kc & 1;
        if (kc + 1 < KCH) {
            int kb1 = (kc + 1) * 32;
            ASYNC_COPY_B128(sel ? ldsd0 : ldsd1,
                            (unsigned long long)(uintptr_t)(gsrc + kb1));
            const unsigned short* abase =
                (kb1 < EP) ? (xrow + kb1) : (hrow + (kb1 - EP));
            a_nxt.q[0] = *(const uint4*)(abase + hi * 16);
            a_nxt.q[1] = *(const uint4*)(abase + hi * 16 + 8);
        }
        FragBF b0, b1, b2, b3;
        const unsigned short* s0 = &smW[sel][ 0 + lo][hi * 16];
        const unsigned short* s1 = &smW[sel][16 + lo][hi * 16];
        const unsigned short* s2 = &smW[sel][32 + lo][hi * 16];
        const unsigned short* s3 = &smW[sel][48 + lo][hi * 16];
        b0.q[0] = *(const uint4*)(s0); b0.q[1] = *(const uint4*)(s0 + 8);
        b1.q[0] = *(const uint4*)(s1); b1.q[1] = *(const uint4*)(s1 + 8);
        b2.q[0] = *(const uint4*)(s2); b2.q[1] = *(const uint4*)(s2 + 8);
        b3.q[0] = *(const uint4*)(s3); b3.q[1] = *(const uint4*)(s3 + 8);
        acc[0] = __builtin_amdgcn_wmma_f32_16x16x32_bf16(false, a_cur.v, false, b0.v, (short)0, acc[0], false, false);
        acc[1] = __builtin_amdgcn_wmma_f32_16x16x32_bf16(false, a_cur.v, false, b1.v, (short)0, acc[1], false, false);
        acc[2] = __builtin_amdgcn_wmma_f32_16x16x32_bf16(false, a_cur.v, false, b2.v, (short)0, acc[2], false, false);
        acc[3] = __builtin_amdgcn_wmma_f32_16x16x32_bf16(false, a_cur.v, false, b3.v, (short)0, acc[3], false, false);
        a_cur = a_nxt;
        WAIT_ASYNC0();
        __syncthreads();
    }
}

// ---------------- recurrent step: gates  (r,u = sigmoid([x,h]Wg + bg)) ------
// GEMM M=128, N=2048, K=1344. 256 waves -> <<<32, 256>>>
__global__ void gru_gates(const unsigned short* __restrict__ xp, int t,
                          const unsigned short* __restrict__ hbf,
                          const float* __restrict__ hf,
                          const unsigned short* __restrict__ Wt,   // [NG][KT]
                          const float* __restrict__ bg,
                          float* __restrict__ u_out,               // [B][H]
                          unsigned short* __restrict__ rh_out) {   // [B][H] bf16
    __shared__ __align__(16) unsigned short smW[2][64][32];
    int wid    = (blockIdx.x * blockDim.x + threadIdx.x) >> 5;
    int lane   = threadIdx.x & 31;
    int lo     = lane & 15, hi = lane >> 4;
    int mtile  = wid & 7;        // 8 M tiles
    int ngroup = wid >> 3;       // 32 N groups of 64 (uniform per block)
    int arow   = mtile * 16 + lo;

    const unsigned short* xrow = xp + ((size_t)t * Bz + arow) * EP;
    const unsigned short* hrow = hbf + (size_t)arow * Hh;

    v8f acc[4] = {v8f{}, v8f{}, v8f{}, v8f{}};
    gru_gemm(xrow, hrow, Wt, ngroup * 64, smW, acc);

    int mbase = mtile * 16 + hi * 8;
    #pragma unroll
    for (int nt = 0; nt < 4; ++nt) {
        int n = ngroup * 64 + nt * 16 + lo;
        float bias = bg[n];
        #pragma unroll
        for (int i = 0; i < 8; ++i) {
            int m = mbase + i;
            float s = 1.0f / (1.0f + __expf(-(acc[nt][i] + bias)));
            if (n < Hh)
                rh_out[(size_t)m * Hh + n] = f2bf(s * hf[(size_t)m * Hh + n]);
            else
                u_out[(size_t)m * Hh + (n - Hh)] = s;
        }
    }
}

// ---------------- recurrent step: candidate + update ------------------------
// c = tanh([x, r*h]Wc + bc);  h' = u*h + (1-u)*c; freeze past seqlen.
// GEMM M=128, N=1024, K=1344. 128 waves -> <<<16, 256>>>
__global__ void gru_update(const unsigned short* __restrict__ xp, int t,
                           const unsigned short* __restrict__ rhbf,
                           const unsigned short* __restrict__ Wt,  // [H][KT]
                           const float* __restrict__ bc,
                           const float* __restrict__ u_in,
                           const int* __restrict__ slen,
                           float* __restrict__ hf,
                           unsigned short* __restrict__ hbf) {
    __shared__ __align__(16) unsigned short smW[2][64][32];
    int wid    = (blockIdx.x * blockDim.x + threadIdx.x) >> 5;
    int lane   = threadIdx.x & 31;
    int lo     = lane & 15, hi = lane >> 4;
    int mtile  = wid & 7;        // 8 M tiles
    int ngroup = wid >> 3;       // 16 N groups of 64 (uniform per block)
    int arow   = mtile * 16 + lo;

    const unsigned short* xrow = xp + ((size_t)t * Bz + arow) * EP;
    const unsigned short* rrow = rhbf + (size_t)arow * Hh;

    v8f acc[4] = {v8f{}, v8f{}, v8f{}, v8f{}};
    gru_gemm(xrow, rrow, Wt, ngroup * 64, smW, acc);

    int mbase = mtile * 16 + hi * 8;
    #pragma unroll
    for (int nt = 0; nt < 4; ++nt) {
        int n = ngroup * 64 + nt * 16 + lo;
        float bias = bc[n];
        #pragma unroll
        for (int i = 0; i < 8; ++i) {
            int m = mbase + i;
            float c  = tanhf(acc[nt][i] + bias);
            float u  = u_in[(size_t)m * Hh + n];
            float ho = hf[(size_t)m * Hh + n];
            float hn = u * ho + (1.0f - u) * c;
            float o  = (t < slen[m]) ? hn : ho;
            hf[(size_t)m * Hh + n]  = o;
            hbf[(size_t)m * Hh + n] = f2bf(o);
        }
    }
}

// ---------------- final linear: [B, 2H] @ [2H, 4] + b -----------------------
__global__ void predict(const float* __restrict__ hh, const float* __restrict__ hb,
                        const float* __restrict__ Wp, const float* __restrict__ bp,
                        float* __restrict__ out) {
    int idx = blockIdx.x * blockDim.x + threadIdx.x;
    if (idx >= Bz * NCLS) return;
    int b = idx >> 2, c = idx & 3;
    float s = bp[c];
    for (int k = 0; k < Hh; ++k) s += hh[(size_t)b * Hh + k] * Wp[(size_t)k * NCLS + c];
    for (int k = 0; k < Hh; ++k) s += hb[(size_t)b * Hh + k] * Wp[(size_t)(Hh + k) * NCLS + c];
    out[idx] = s;
}

// ---------------------------------------------------------------------------

extern "C" void kernel_launch(void* const* d_in, const int* in_sizes, int n_in,
                              void* d_out, int out_size, void* d_ws, size_t ws_size,
                              hipStream_t stream) {
    const int*   headlines = (const int*)d_in[0];
    const int*   bodies    = (const int*)d_in[1];
    const float* emb       = (const float*)d_in[2];
    const float* Wg_h = (const float*)d_in[3];
    const float* bg_h = (const float*)d_in[4];
    const float* Wc_h = (const float*)d_in[5];
    const float* bc_h = (const float*)d_in[6];
    const float* Wg_b = (const float*)d_in[7];
    const float* bg_b = (const float*)d_in[8];
    const float* Wc_b = (const float*)d_in[9];
    const float* bc_b = (const float*)d_in[10];
    const float* Wp   = (const float*)d_in[11];
    const float* bp   = (const float*)d_in[12];
    float* out = (float*)d_out;

    // workspace carve-out (~85 MB total)
    char* ws = (char*)d_ws;
    size_t off = 0;
    auto carve = [&](size_t bytes) {
        off = (off + 255) & ~(size_t)255;
        void* p = ws + off;
        off += bytes;
        return p;
    };
    unsigned short* WgHt = (unsigned short*)carve((size_t)NG * KT * 2);
    unsigned short* WcHt = (unsigned short*)carve((size_t)Hh * KT * 2);
    unsigned short* WgBt = (unsigned short*)carve((size_t)NG * KT * 2);
    unsigned short* WcBt = (unsigned short*)carve((size_t)Hh * KT * 2);
    unsigned short* xh   = (unsigned short*)carve((size_t)HLEN * Bz * EP * 2);
    unsigned short* xb   = (unsigned short*)carve((size_t)BLEN * Bz * EP * 2);
    float*          hH   = (float*)carve((size_t)Bz * Hh * 4);
    float*          hB   = (float*)carve((size_t)Bz * Hh * 4);
    unsigned short* hbf  = (unsigned short*)carve((size_t)Bz * Hh * 2);
    float*          ubuf = (float*)carve((size_t)Bz * Hh * 4);
    unsigned short* rhbf = (unsigned short*)carve((size_t)Bz * Hh * 2);
    int*            used = (int*)carve((size_t)Bz * BLEN * 4);
    int*            slen = (int*)carve((size_t)Bz * 4);
    (void)ws_size; (void)in_sizes; (void)n_in; (void)out_size;

    // bf16 transposed/padded weights (resident in 192MB L2 across all steps)
    convert_weight<<<(NG * KT + 255) / 256, 256, 0, stream>>>(Wg_h, WgHt, NG);
    convert_weight<<<(Hh * KT + 255) / 256, 256, 0, stream>>>(Wc_h, WcHt, Hh);
    convert_weight<<<(NG * KT + 255) / 256, 256, 0, stream>>>(Wg_b, WgBt, NG);
    convert_weight<<<(Hh * KT + 255) / 256, 256, 0, stream>>>(Wc_b, WcBt, Hh);

    auto run_seq = [&](const int* toks, int T, unsigned short* xp,
                       const unsigned short* Wgt, const float* bg,
                       const unsigned short* Wct, const float* bc, float* hf) {
        int total = T * Bz * EP;
        gather_embed<<<(total + 255) / 256, 256, 0, stream>>>(toks, T, emb, xp, total);
        seq_used<<<(Bz * T + 255) / 256, 256, 0, stream>>>(toks, T, emb, used);
        seq_len<<<1, Bz, 0, stream>>>(used, T, slen);
        zero_state<<<(Bz * Hh + 255) / 256, 256, 0, stream>>>(hf, hbf);
        for (int t = 0; t < T; ++t) {
            gru_gates<<<32, 256, 0, stream>>>(xp, t, hbf, hf, Wgt, bg, ubuf, rhbf);
            gru_update<<<16, 256, 0, stream>>>(xp, t, rhbf, Wct, bc, ubuf, slen, hf, hbf);
        }
    };

    run_seq(headlines, HLEN, xh, WgHt, bg_h, WcHt, bc_h, hH);
    run_seq(bodies,    BLEN, xb, WgBt, bg_b, WcBt, bc_b, hB);

    predict<<<2, 256, 0, stream>>>(hH, hB, Wp, bp, out);
}
